// PN2S2CNNCls_52501680226351
// MI455X (gfx1250) — matrix operand order repaired and checked by reading, not compiled, not run on hardware
//
#include <hip/hip_runtime.h>
#include <hip/hip_bf16.h>

#define B_SZ   16
#define N_PTS  4096
#define M_PTS  512
#define K_GRP  4
#define KNN_K  4
#define C_F    67
#define RAD    0.2f
#define NEGV   -1000000000.0f

typedef __attribute__((ext_vector_type(16))) _Float16 v16h;
typedef __attribute__((ext_vector_type(8)))  _Float16 v8h;
typedef __attribute__((ext_vector_type(8)))  float    v8f;

// ---------------------------------------------------------------------------
// WMMA GEMM on pre-packed f16 operands.
//   O[r][c] = act( sum_k A[r][k] * W[c][k] + bias[c] )
// A: [Mr x Kp] f16 (lda = Kp multiple of 32, zero-padded), W: [Nc x Kp] f16.
// Mr, Nc multiples of 16. One wave -> one 16x16 tile. Per k-step: 4 coalesced
// 16B loads (global_load_b128) + 1 v_wmma_f32_16x16x32_f16. f32 accumulate.
// ---------------------------------------------------------------------------
__device__ __forceinline__ v16h frag_cat(v8h lo, v8h hi) {
  v16h r;
  #pragma unroll
  for (int i = 0; i < 8; ++i) { r[i] = lo[i]; r[i + 8] = hi[i]; }
  return r;
}

__device__ __forceinline__ v8f wmma_step(const _Float16* __restrict__ Arow,
                                         const _Float16* __restrict__ Wrow,
                                         int k0, int half, v8f c)
{
  // A 16x32 layout: lanes 0-15 hold K {0..7,16..23}; lanes 16-31 {8..15,24..31}
  v8h alo = *(const v8h*)(Arow + k0 + half * 8);
  v8h ahi = *(const v8h*)(Arow + k0 + 16 + half * 8);
  // B 32x16 layout: lanes 0-15 hold K {0..7,16..23}; lanes 16-31 {8..15,24..31}
  v8h blo = *(const v8h*)(Wrow + k0 + half * 8);
  v8h bhi = *(const v8h*)(Wrow + k0 + 16 + half * 8);
  return __builtin_amdgcn_wmma_f32_16x16x32_f16(false, frag_cat(alo, ahi),
                                                false, frag_cat(blo, bhi),
                                                (short)0, c, false, false);
}

__global__ __launch_bounds__(128) void wmma_gemm_kernel(
    const _Float16* __restrict__ A, long long strideA,
    const _Float16* __restrict__ W, long long strideW,
    const float* __restrict__ bias,
    float* __restrict__ O, long long strideO,
    _Float16* __restrict__ Oh, long long strideOh,
    int Mr, int Nc, int Kp, int ldah, int ldwh, int ldo, int ldoh, int relu)
{
  int batch = blockIdx.y;
  A += (size_t)batch * strideA;
  W += (size_t)batch * strideW;

  int tjn = Nc >> 4;
  int wave = blockIdx.x * (blockDim.x >> 5) + (threadIdx.x >> 5);
  if (wave >= (Mr >> 4) * tjn) return;       // wave-uniform exit, EXEC stays full
  int ti = wave / tjn, tj = wave % tjn;
  int i0 = ti << 4, j0 = tj << 4;

  int lane = threadIdx.x & 31;
  int r    = lane & 15;
  int half = lane >> 4;
  const _Float16* Arow = A + (size_t)(i0 + r) * ldah;
  const _Float16* Wrow = W + (size_t)(j0 + r) * ldwh;

  v8f c = {};
  int k0 = 0;
  for (; k0 + 64 <= Kp; k0 += 64) {
    if (k0 + 64 < Kp) __builtin_prefetch(Arow + k0 + 64, 0, 1); // global_prefetch_b8
    c = wmma_step(Arow, Wrow, k0,      half, c);
    c = wmma_step(Arow, Wrow, k0 + 32, half, c);
  }
  if (k0 < Kp)
    c = wmma_step(Arow, Wrow, k0, half, c);

  // C/D layout: lane col = j0 + (lane&15); rows = i0 + half*8 + v
  int col = j0 + r;
  float bv = bias ? bias[col] : 0.0f;
  #pragma unroll
  for (int v = 0; v < 8; ++v) {
    int row = i0 + half * 8 + v;
    float val = c[v] + bv;
    if (relu) val = val > 0.0f ? val : 0.0f;
    if (O)  O [(size_t)batch * strideO  + (size_t)row * ldo  + col] = val;
    if (Oh) Oh[(size_t)batch * strideOh + (size_t)row * ldoh + col] = (_Float16)val;
  }
}

// ---------------------------------------------------------------------------
// f16 packing of weights / biases (zero-padded rows & K)
// ---------------------------------------------------------------------------
__global__ void pack_f16_kernel(const float* __restrict__ src, _Float16* __restrict__ dst,
                                int rows, int cols, int rpad, int ldd)
{
  int id = blockIdx.x * blockDim.x + threadIdx.x;
  if (id >= rpad * ldd) return;
  int rr = id / ldd, cc = id % ldd;
  float v = (rr < rows && cc < cols) ? src[(size_t)rr * cols + cc] : 0.0f;
  dst[id] = (_Float16)v;
}

__global__ void pack_bias_kernel(const float* __restrict__ src, float* __restrict__ dst,
                                 int n, int npad)
{
  int id = blockIdx.x * blockDim.x + threadIdx.x;
  if (id >= npad) return;
  dst[id] = (id < n) ? src[id] : 0.0f;
}

// ---------------------------------------------------------------------------
// Farthest point sampling: one block per batch, 511 serialized argmax rounds.
// ---------------------------------------------------------------------------
__global__ __launch_bounds__(256) void fps_kernel(const float* __restrict__ pts,
                                                  int* __restrict__ idx_out)
{
  int b = blockIdx.x;
  const float* px = pts + (size_t)b * 3 * N_PTS;   // [3][N] planes
  __shared__ float mind[N_PTS];
  __shared__ float rv[256];
  __shared__ int   ri[256];
  __shared__ float selpt[3];
  int t = threadIdx.x;

  float x0 = px[0], y0 = px[N_PTS], z0 = px[2 * N_PTS];
  for (int n = t; n < N_PTS; n += 256) {
    float dx = px[n] - x0, dy = px[N_PTS + n] - y0, dz = px[2 * N_PTS + n] - z0;
    mind[n] = dx * dx + dy * dy + dz * dz;
  }
  if (t == 0) idx_out[b * M_PTS] = 0;
  __syncthreads();

  for (int s = 1; s < M_PTS; ++s) {
    float bv = -1.0f; int bi = 0;
    for (int n = t; n < N_PTS; n += 256) {
      float v = mind[n];
      if (v > bv) { bv = v; bi = n; }
    }
    rv[t] = bv; ri[t] = bi;
    __syncthreads();
    for (int off = 128; off > 0; off >>= 1) {
      if (t < off) {
        float v2 = rv[t + off]; int i2 = ri[t + off];
        if (v2 > rv[t] || (v2 == rv[t] && i2 < ri[t])) { rv[t] = v2; ri[t] = i2; }
      }
      __syncthreads();
    }
    if (t == 0) {
      int sel = ri[0];
      idx_out[b * M_PTS + s] = sel;
      selpt[0] = px[sel]; selpt[1] = px[N_PTS + sel]; selpt[2] = px[2 * N_PTS + sel];
    }
    __syncthreads();
    float sx = selpt[0], sy = selpt[1], sz = selpt[2];
    for (int n = t; n < N_PTS; n += 256) {
      float dx = px[n] - sx, dy = px[N_PTS + n] - sy, dz = px[2 * N_PTS + n] - sz;
      float d = dx * dx + dy * dy + dz * dz;
      if (d < mind[n]) mind[n] = d;
    }
    __syncthreads();
  }
}

__global__ void gather_newpts_kernel(const float* __restrict__ pts,
                                     const int* __restrict__ idx,
                                     float* __restrict__ np_)
{
  int i = blockIdx.x * blockDim.x + threadIdx.x;
  if (i >= B_SZ * M_PTS) return;
  int b = i / M_PTS;
  int n = idx[i];
  const float* px = pts + (size_t)b * 3 * N_PTS;
  np_[(size_t)i * 3 + 0] = px[n];
  np_[(size_t)i * 3 + 1] = px[N_PTS + n];
  np_[(size_t)i * 3 + 2] = px[2 * N_PTS + n];
}

// ---------------------------------------------------------------------------
// Ball query (first-K-by-index within radius) + grouped MLP 3->32->64 +
// masked max over group. One block per (b,m) query point.
// ---------------------------------------------------------------------------
__global__ __launch_bounds__(128) void ballquery_mlp_kernel(
    const float* __restrict__ pts, const float* __restrict__ newpts,
    const float* __restrict__ w1 /*[32x3]*/, const float* __restrict__ w2 /*[64x32]*/,
    float* __restrict__ localf /*[B*M,64]*/)
{
  int bm = blockIdx.x;
  int b = bm / M_PTS;
  const float* px = pts + (size_t)b * 3 * N_PTS;
  float cx = newpts[(size_t)bm * 3 + 0];
  float cy = newpts[(size_t)bm * 3 + 1];
  float cz = newpts[(size_t)bm * 3 + 2];

  __shared__ int   scnt[128];
  __shared__ int   shits[128 * K_GRP];
  __shared__ int   sel[K_GRP];
  __shared__ int   scount;
  __shared__ float sgrp[K_GRP * 3];
  __shared__ float sh1[K_GRP * 32];
  __shared__ float sh2[K_GRP * 64];

  int t = threadIdx.x;
  const float r2 = RAD * RAD;
  int base = t * 32;
  int c = 0; int h[K_GRP];
  for (int j = 0; j < 32; ++j) {
    int n = base + j;
    float dx = px[n] - cx, dy = px[N_PTS + n] - cy, dz = px[2 * N_PTS + n] - cz;
    float d2 = dx * dx + dy * dy + dz * dz;
    if (d2 < r2) { if (c < K_GRP) h[c] = n; c++; }
  }
  scnt[t] = c;
  for (int j = 0; j < K_GRP; ++j) shits[t * K_GRP + j] = (j < c) ? h[j] : -1;
  __syncthreads();

  if (t == 0) {
    int tot = 0, k = 0;
    for (int u = 0; u < 128; ++u) {
      int cc = scnt[u];
      int take = cc < K_GRP ? cc : K_GRP;
      for (int j = 0; j < take && k < K_GRP; ++j) sel[k++] = shits[u * K_GRP + j];
      tot += cc;
    }
    int first = (k > 0) ? sel[0] : (N_PTS - 1);   // reference fallback -> n-1
    for (int j = k; j < K_GRP; ++j) sel[j] = first;
    scount = tot < K_GRP ? tot : K_GRP;
  }
  __syncthreads();

  if (t < K_GRP * 3) {
    int kk = t / 3, cc = t % 3;
    int n = sel[kk];
    float v   = (cc == 0) ? px[n] : (cc == 1 ? px[N_PTS + n] : px[2 * N_PTS + n]);
    float ctr = (cc == 0) ? cx    : (cc == 1 ? cy            : cz);
    sgrp[kk * 3 + cc] = (v - ctr) / RAD;
  }
  __syncthreads();

  { // stage 1: [K,3] x w1^T -> [K,32], relu
    int kk = t >> 5, o = t & 31;
    float s = 0.0f;
    for (int i = 0; i < 3; ++i) s += sgrp[kk * 3 + i] * w1[o * 3 + i];
    sh1[kk * 32 + o] = s > 0.0f ? s : 0.0f;
  }
  __syncthreads();

  for (int id = t; id < K_GRP * 64; id += 128) { // stage 2: [K,32] x w2^T -> [K,64]
    int kk = id >> 6, o = id & 63;
    float s = 0.0f;
    for (int i = 0; i < 32; ++i) s += sh1[kk * 32 + i] * w2[o * 32 + i];
    s = s > 0.0f ? s : 0.0f;
    sh2[id] = (kk < scount) ? s : NEGV;           // mask invalid group members
  }
  __syncthreads();

  if (t < 64) {
    float m = sh2[t];
    for (int kk = 1; kk < K_GRP; ++kk) { float v = sh2[kk * 64 + t]; m = v > m ? v : m; }
    localf[(size_t)bm * 64 + t] = m;
  }
}

// F (f32, ld 67) and F16 (f16, ld 96 zero-padded): [local(64) | xyz(3) | 0-pad]
__global__ void build_f_kernel(const float* __restrict__ localf,
                               const float* __restrict__ newpts,
                               float* __restrict__ F, _Float16* __restrict__ F16)
{
  int id = blockIdx.x * blockDim.x + threadIdx.x;
  if (id >= B_SZ * M_PTS * 96) return;
  int bm = id / 96, c = id % 96;
  float v = 0.0f;
  if (c < 64)      v = localf[(size_t)bm * 64 + c];
  else if (c < 67) v = newpts[(size_t)bm * 3 + (c - 64)];
  F16[id] = (_Float16)v;
  if (c < C_F) F[(size_t)bm * C_F + c] = v;
}

__global__ void diag_kernel(const float* __restrict__ G, float* __restrict__ sq)
{
  int id = blockIdx.x * blockDim.x + threadIdx.x;
  if (id >= B_SZ * M_PTS) return;
  int b = id / M_PTS, i = id % M_PTS;
  sq[id] = G[(size_t)b * M_PTS * M_PTS + (size_t)i * M_PTS + i];
}

__global__ void knn_kernel(const float* __restrict__ G, const float* __restrict__ sq,
                           int* __restrict__ nn)
{
  int id = blockIdx.x * blockDim.x + threadIdx.x;
  if (id >= B_SZ * M_PTS) return;
  int b = id / M_PTS, i = id % M_PTS;
  const float* Gi  = G + ((size_t)b * M_PTS + i) * M_PTS;
  const float* sqb = sq + (size_t)b * M_PTS;
  float dii = sqb[i];
  float bv[KNN_K]; int bi[KNN_K]; int filled = 0;
  for (int j = 0; j < M_PTS; ++j) {
    float d = dii + sqb[j] - 2.0f * Gi[j];
    if (filled < KNN_K) {
      int p = filled;
      while (p > 0 && d < bv[p - 1]) { bv[p] = bv[p - 1]; bi[p] = bi[p - 1]; --p; }
      bv[p] = d; bi[p] = j; ++filled;
    } else if (d < bv[KNN_K - 1]) {
      int p = KNN_K - 1;
      while (p > 0 && d < bv[p - 1]) { bv[p] = bv[p - 1]; bi[p] = bi[p - 1]; --p; }
      bv[p] = d; bi[p] = j;
    }
  }
  for (int k = 0; k < KNN_K; ++k) nn[(size_t)id * KNN_K + k] = bi[k];
}

// Edge features -> f16 [B*M*KNN x 160] (134 valid + zero pad)
__global__ void edge_build_kernel(const float* __restrict__ F,
                                  const int* __restrict__ nn,
                                  _Float16* __restrict__ E)
{
  int id = blockIdx.x * blockDim.x + threadIdx.x;
  if (id >= B_SZ * M_PTS * KNN_K * 160) return;
  int c    = id % 160;
  int rest = id / 160;
  int kk   = rest % KNN_K;
  int bm   = rest / KNN_K;
  int b    = bm / M_PTS;
  float v = 0.0f;
  if (c < C_F) {
    v = F[(size_t)bm * C_F + c];
  } else if (c < 2 * C_F) {
    int j = nn[(size_t)bm * KNN_K + kk];
    int cc = c - C_F;
    v = F[((size_t)b * M_PTS + j) * C_F + cc] - F[(size_t)bm * C_F + cc];
  }
  E[id] = (_Float16)v;
}

__global__ void maxk_kernel(const float* __restrict__ in, _Float16* __restrict__ out, int C)
{
  int id = blockIdx.x * blockDim.x + threadIdx.x;
  if (id >= B_SZ * M_PTS * C) return;
  int c = id % C, p = id / C;
  float m = in[((size_t)p * KNN_K) * C + c];
  for (int kk = 1; kk < KNN_K; ++kk) {
    float v = in[((size_t)p * KNN_K + kk) * C + c];
    m = v > m ? v : m;
  }
  out[id] = (_Float16)m;
}

__global__ void maxm_kernel(const float* __restrict__ in, _Float16* __restrict__ out, int C)
{
  int id = blockIdx.x * blockDim.x + threadIdx.x;
  if (id >= B_SZ * C) return;
  int b = id / C, c = id % C;
  float m = in[((size_t)b * M_PTS) * C + c];
  for (int mm = 1; mm < M_PTS; ++mm) {
    float v = in[((size_t)b * M_PTS + mm) * C + c];
    m = v > m ? v : m;
  }
  out[id] = (_Float16)m;
}

__global__ void trans_kernel(const float* __restrict__ t9 /*[16x16]*/,
                             float* __restrict__ trans /*[B,9]*/)
{
  int id = blockIdx.x * blockDim.x + threadIdx.x;
  if (id >= B_SZ * 9) return;
  int b = id / 9, e = id % 9;
  trans[id] = t9[(size_t)b * 16 + e] + ((e % 4 == 0) ? 1.0f : 0.0f);
}

__global__ void xyzt_kernel(const float* __restrict__ trans,
                            const float* __restrict__ newpts,
                            float* __restrict__ xyzt)
{
  int id = blockIdx.x * blockDim.x + threadIdx.x;
  if (id >= B_SZ * M_PTS) return;
  int b = id / M_PTS;
  for (int i = 0; i < 3; ++i) {
    float s = 0.0f;
    for (int j = 0; j < 3; ++j)
      s += trans[(size_t)b * 9 + i * 3 + j] * newpts[(size_t)id * 3 + j];
    xyzt[(size_t)id * 3 + i] = s;
  }
}

// X1 f16 [B*M x 96]: [local(64) | xyz(3) | xyz_t(3) | 0-pad]
__global__ void x1_build_kernel(const float* __restrict__ localf,
                                const float* __restrict__ newpts,
                                const float* __restrict__ xyzt,
                                _Float16* __restrict__ X1)
{
  int id = blockIdx.x * blockDim.x + threadIdx.x;
  if (id >= B_SZ * M_PTS * 96) return;
  int bm = id / 96, c = id % 96;
  float v = 0.0f;
  if (c < 64)      v = localf[(size_t)bm * 64 + c];
  else if (c < 67) v = newpts[(size_t)bm * 3 + (c - 64)];
  else if (c < 70) v = xyzt[(size_t)bm * 3 + (c - 67)];
  X1[id] = (_Float16)v;
}

__global__ void out_copy_kernel(const float* __restrict__ opad, float* __restrict__ out)
{
  int id = blockIdx.x * blockDim.x + threadIdx.x;
  if (id >= B_SZ * 40) return;
  int b = id / 40, c = id % 40;
  out[id] = opad[(size_t)b * 48 + c];
}

// ---------------------------------------------------------------------------
static inline int cdiv(long long a, long long b) { return (int)((a + b - 1) / b); }

extern "C" void kernel_launch(void* const* d_in, const int* in_sizes, int n_in,
                              void* d_out, int out_size, void* d_ws, size_t ws_size,
                              hipStream_t stream)
{
  (void)in_sizes; (void)n_in; (void)out_size; (void)ws_size;
  const float* points  = (const float*)d_in[0];
  const float* s2_w1   = (const float*)d_in[1];
  const float* s2_w2   = (const float*)d_in[2];
  const float* ec_w1   = (const float*)d_in[3];
  const float* ec_w2   = (const float*)d_in[4];
  const float* local_w = (const float*)d_in[5];
  const float* tg_w1   = (const float*)d_in[6];
  const float* tg_b1   = (const float*)d_in[7];
  const float* tg_w2   = (const float*)d_in[8];
  const float* tg_b2   = (const float*)d_in[9];
  const float* lin_w   = (const float*)d_in[10];
  const float* lin_b   = (const float*)d_in[11];
  const float* mlp_w1  = (const float*)d_in[12];
  const float* mlp_w2  = (const float*)d_in[13];
  const float* mlp_w3  = (const float*)d_in[14];
  const float* g_w1    = (const float*)d_in[15];
  const float* g_b1    = (const float*)d_in[16];
  const float* g_w2    = (const float*)d_in[17];
  const float* g_b2    = (const float*)d_in[18];
  const float* cls_w   = (const float*)d_in[19];
  const float* cls_b   = (const float*)d_in[20];

  char* ws = (char*)d_ws;
  size_t off = 0;
  auto alloc = [&](size_t bytes) -> void* {
    void* p = ws + off;
    off += (bytes + 255) & ~(size_t)255;
    return p;
  };

  const size_t BM = (size_t)B_SZ * M_PTS;
  const size_t BMK = BM * KNN_K;

  // f32 intermediates
  int*      fps_idx = (int*)      alloc(BM * sizeof(int));
  float*    newpts  = (float*)    alloc(BM * 3 * sizeof(float));
  float*    localf  = (float*)    alloc(BM * 64 * sizeof(float));
  float*    F       = (float*)    alloc(BM * C_F * sizeof(float));
  float*    G       = (float*)    alloc(BM * M_PTS * sizeof(float));
  float*    sq      = (float*)    alloc(BM * sizeof(float));
  int*      nn      = (int*)      alloc(BM * KNN_K * sizeof(int));
  float*    H2      = (float*)    alloc(BMK * 128 * sizeof(float));
  float*    HLOC    = (float*)    alloc(BM * 1024 * sizeof(float));
  float*    T9      = (float*)    alloc((size_t)16 * 16 * sizeof(float));
  float*    TRANS   = (float*)    alloc((size_t)B_SZ * 9 * sizeof(float));
  float*    XYZT    = (float*)    alloc(BM * 3 * sizeof(float));
  float*    Y3      = (float*)    alloc(BM * 1024 * sizeof(float));
  float*    OUTPAD  = (float*)    alloc((size_t)B_SZ * 48 * sizeof(float));
  float*    lin_bp  = (float*)    alloc(16 * sizeof(float));
  float*    cls_bp  = (float*)    alloc(48 * sizeof(float));

  // f16 activations (K padded to multiples of 32, zero-filled)
  _Float16* F16   = (_Float16*) alloc(BM * 96 * sizeof(_Float16));
  _Float16* Eh    = (_Float16*) alloc(BMK * 160 * sizeof(_Float16));
  _Float16* H1h   = (_Float16*) alloc(BMK * 64 * sizeof(_Float16));
  _Float16* HMAXh = (_Float16*) alloc(BM * 128 * sizeof(_Float16));
  _Float16* TGh   = (_Float16*) alloc((size_t)B_SZ * 1024 * sizeof(_Float16));
  _Float16* T1h   = (_Float16*) alloc((size_t)B_SZ * 512 * sizeof(_Float16));
  _Float16* T2h   = (_Float16*) alloc((size_t)B_SZ * 256 * sizeof(_Float16));
  _Float16* X1h   = (_Float16*) alloc(BM * 96 * sizeof(_Float16));
  _Float16* Y1h   = (_Float16*) alloc(BM * 256 * sizeof(_Float16));
  _Float16* Y2h   = (_Float16*) alloc(BM * 512 * sizeof(_Float16));
  _Float16* GMAXh = (_Float16*) alloc((size_t)B_SZ * 1024 * sizeof(_Float16));
  _Float16* Z1h   = (_Float16*) alloc((size_t)B_SZ * 512 * sizeof(_Float16));
  _Float16* Z2h   = (_Float16*) alloc((size_t)B_SZ * 256 * sizeof(_Float16));

  // f16 weights (rows padded to mult of 16, K padded to mult of 32)
  _Float16* ec_w1h   = (_Float16*) alloc((size_t)64 * 160 * sizeof(_Float16));
  _Float16* ec_w2h   = (_Float16*) alloc((size_t)128 * 64 * sizeof(_Float16));
  _Float16* local_wh = (_Float16*) alloc((size_t)1024 * 128 * sizeof(_Float16));
  _Float16* tg_w1h   = (_Float16*) alloc((size_t)512 * 1024 * sizeof(_Float16));
  _Float16* tg_w2h   = (_Float16*) alloc((size_t)256 * 512 * sizeof(_Float16));
  _Float16* lin_wh   = (_Float16*) alloc((size_t)16 * 256 * sizeof(_Float16));
  _Float16* mlp_w1h  = (_Float16*) alloc((size_t)256 * 96 * sizeof(_Float16));
  _Float16* mlp_w2h  = (_Float16*) alloc((size_t)512 * 256 * sizeof(_Float16));
  _Float16* mlp_w3h  = (_Float16*) alloc((size_t)1024 * 512 * sizeof(_Float16));
  _Float16* g_w1h    = (_Float16*) alloc((size_t)512 * 1024 * sizeof(_Float16));
  _Float16* g_w2h    = (_Float16*) alloc((size_t)256 * 512 * sizeof(_Float16));
  _Float16* cls_wh   = (_Float16*) alloc((size_t)48 * 256 * sizeof(_Float16));

  auto pack = [&](const float* src, _Float16* dst, int rows, int cols, int rpad, int ldd) {
    pack_f16_kernel<<<cdiv((long long)rpad * ldd, 256), 256, 0, stream>>>(src, dst, rows, cols, rpad, ldd);
  };
  auto gemm = [&](const _Float16* A, long long sA, const _Float16* W, long long sW,
                  const float* bias, float* O, long long sO, _Float16* Oh, long long sOh,
                  int Mr, int Nc, int Kp, int ldah, int ldwh, int ldo, int ldoh,
                  int relu, int nbatch) {
    int tiles = (Mr / 16) * (Nc / 16);
    dim3 grid((tiles + 3) / 4, nbatch);
    wmma_gemm_kernel<<<grid, 128, 0, stream>>>(A, sA, W, sW, bias, O, sO, Oh, sOh,
                                               Mr, Nc, Kp, ldah, ldwh, ldo, ldoh, relu);
  };

  // 0) weight packing (independent of data path)
  pack(ec_w1,   ec_w1h,   64,   134,  64,   160);
  pack(ec_w2,   ec_w2h,   128,  64,   128,  64);
  pack(local_w, local_wh, 1024, 128,  1024, 128);
  pack(tg_w1,   tg_w1h,   512,  1024, 512,  1024);
  pack(tg_w2,   tg_w2h,   256,  512,  256,  512);
  pack(lin_w,   lin_wh,   9,    256,  16,   256);
  pack(mlp_w1,  mlp_w1h,  256,  70,   256,  96);
  pack(mlp_w2,  mlp_w2h,  512,  256,  512,  256);
  pack(mlp_w3,  mlp_w3h,  1024, 512,  1024, 512);
  pack(g_w1,    g_w1h,    512,  1024, 512,  1024);
  pack(g_w2,    g_w2h,    256,  512,  256,  512);
  pack(cls_w,   cls_wh,   40,   256,  48,   256);
  pack_bias_kernel<<<1, 64, 0, stream>>>(lin_b, lin_bp, 9, 16);
  pack_bias_kernel<<<1, 64, 0, stream>>>(cls_b, cls_bp, 40, 48);

  // 1) FPS + gather centers
  fps_kernel<<<B_SZ, 256, 0, stream>>>(points, fps_idx);
  gather_newpts_kernel<<<cdiv(BM, 256), 256, 0, stream>>>(points, fps_idx, newpts);

  // 2) Ball query + grouped MLP -> local features
  ballquery_mlp_kernel<<<(int)BM, 128, 0, stream>>>(points, newpts, s2_w1, s2_w2, localf);

  // 3) TNet features F (f32 + padded f16)
  build_f_kernel<<<cdiv(BM * 96, 256), 256, 0, stream>>>(localf, newpts, F, F16);

  // 4) Batched Gram G[b] = F[b] @ F[b]^T  (WMMA) -> KNN
  gemm(F16, (long long)M_PTS * 96, F16, (long long)M_PTS * 96, nullptr,
       G, (long long)M_PTS * M_PTS, nullptr, 0,
       M_PTS, M_PTS, 96, 96, 96, M_PTS, 0, 0, B_SZ);
  diag_kernel<<<cdiv(BM, 256), 256, 0, stream>>>(G, sq);
  knn_kernel<<<cdiv(BM, 128), 128, 0, stream>>>(G, sq, nn);

  // 5) Edge conv 134->64->128, max over k
  edge_build_kernel<<<cdiv(BMK * 160, 256), 256, 0, stream>>>(F, nn, Eh);
  gemm(Eh, 0, ec_w1h, 0, nullptr, nullptr, 0, H1h, 0,
       (int)BMK, 64, 160, 160, 160, 0, 64, 1, 1);
  gemm(H1h, 0, ec_w2h, 0, nullptr, H2, 0, nullptr, 0,
       (int)BMK, 128, 64, 64, 64, 128, 0, 1, 1);
  maxk_kernel<<<cdiv(BM * 128, 256), 256, 0, stream>>>(H2, HMAXh, 128);

  // 6) TNet local 128->1024, max over M, FC 1024->512->256->9(16)
  gemm(HMAXh, 0, local_wh, 0, nullptr, HLOC, 0, nullptr, 0,
       (int)BM, 1024, 128, 128, 128, 1024, 0, 1, 1);
  maxm_kernel<<<cdiv((size_t)B_SZ * 1024, 256), 256, 0, stream>>>(HLOC, TGh, 1024);
  gemm(TGh, 0, tg_w1h, 0, tg_b1, nullptr, 0, T1h, 0, 16, 512, 1024, 1024, 1024, 0, 512, 1, 1);
  gemm(T1h, 0, tg_w2h, 0, tg_b2, nullptr, 0, T2h, 0, 16, 256, 512, 512, 512, 0, 256, 1, 1);
  gemm(T2h, 0, lin_wh, 0, lin_bp, T9, 0, nullptr, 0, 16, 16, 256, 256, 256, 16, 0, 0, 1);
  trans_kernel<<<1, 256, 0, stream>>>(T9, TRANS);
  xyzt_kernel<<<cdiv(BM, 256), 256, 0, stream>>>(TRANS, newpts, XYZT);

  // 7) Main MLP 70->256->512->1024, global max
  x1_build_kernel<<<cdiv(BM * 96, 256), 256, 0, stream>>>(localf, newpts, XYZT, X1h);
  gemm(X1h, 0, mlp_w1h, 0, nullptr, nullptr, 0, Y1h, 0, (int)BM, 256, 96, 96, 96, 0, 256, 1, 1);
  gemm(Y1h, 0, mlp_w2h, 0, nullptr, nullptr, 0, Y2h, 0, (int)BM, 512, 256, 256, 256, 0, 512, 1, 1);
  gemm(Y2h, 0, mlp_w3h, 0, nullptr, Y3, 0, nullptr, 0, (int)BM, 1024, 512, 512, 512, 1024, 0, 1, 1);
  maxm_kernel<<<cdiv((size_t)B_SZ * 1024, 256), 256, 0, stream>>>(Y3, GMAXh, 1024);

  // 8) Head 1024->512->256->40(48)
  gemm(GMAXh, 0, g_w1h, 0, g_b1, nullptr, 0, Z1h, 0, 16, 512, 1024, 1024, 1024, 0, 512, 1, 1);
  gemm(Z1h, 0, g_w2h, 0, g_b2, nullptr, 0, Z2h, 0, 16, 256, 512, 512, 512, 0, 256, 1, 1);
  gemm(Z2h, 0, cls_wh, 0, cls_bp, OUTPAD, 0, nullptr, 0, 16, 48, 256, 256, 256, 48, 0, 0, 1);
  out_copy_kernel<<<cdiv((size_t)B_SZ * 40, 256), 256, 0, stream>>>(OUTPAD, (float*)d_out);
}
